// WSDModel_15771119911424
// MI455X (gfx1250) — matrix operand order repaired and checked by reading, not compiled, not run on hardware
//
#include <hip/hip_runtime.h>
#include <math.h>

// ---------------------------------------------------------------------------
// WSD pipeline for MI455X (gfx1250, wave32, WMMA).
//   embed gather -> [per pass: xproj big GEMM (WMMA) -> 128x (gates GEMM (WMMA)
//   + LSTM cell pointwise)] x 4 passes -> Wout head GEMM (WMMA) -> expert GEMV.
// GEMM inner loop: 32x64 wave tile (MT=2 x NT=4), double-buffered K pipeline
// (fragments for step k+32 loaded before WMMAs of step k), unconditional b128
// fragment loads (weights pre-padded), 8 v_wmma per K-step.
// __launch_bounds__(256, 2): trade occupancy for registers so the ~190-VGPR
// pipelined loop does not spill to scratch (latency is hidden by the double
// buffer, not by wave count).
// ---------------------------------------------------------------------------

typedef __attribute__((ext_vector_type(16))) _Float16 v16h;
typedef __attribute__((ext_vector_type(8)))  _Float16 v8h;
typedef __attribute__((ext_vector_type(8)))  float    v8f;

namespace {

constexpr int Bz   = 64;
constexpr int Sq   = 128;
constexpr int Ed   = 300;
constexpr int EP   = 320;    // E padded to multiple of 64 (K pipeline step)
constexpr int EdP  = 320;    // Wout N=300 padded to multiple of 64 (tile width)
constexpr int Hh   = 512;
constexpr int TwoH = 1024;
constexpr int G4   = 2048;   // 4*H gate width
constexpr int NTOK = Bz * Sq;
constexpr int NS   = 16;     // NSYN

// ---- WMMA fragment loaders (16x16x32 f16, wave32; ISA 7.12.2 layouts) ------

__device__ __forceinline__ v16h load_a_frag(const _Float16* __restrict__ A,
                                            int lda, int mbase, int kbase,
                                            int lane) {
  // A: row-major [M, lda]. Lane holds row mbase+(lane&15);
  // elems 0..7 -> K = k0..k0+7, elems 8..15 -> K = k0+16..k0+23,
  // k0 = kbase + (lane>=16 ? 8 : 0).
  int row = mbase + (lane & 15);
  int k0  = kbase + ((lane & 16) ? 8 : 0);
  const v8h* p0 = reinterpret_cast<const v8h*>(A + (size_t)row * lda + k0);
  const v8h* p1 = reinterpret_cast<const v8h*>(A + (size_t)row * lda + k0 + 16);
  v8h a = *p0, b = *p1;
  v16h r;
#pragma unroll
  for (int i = 0; i < 8; ++i) { r[i] = a[i]; r[i + 8] = b[i]; }
  return r;
}

__device__ __forceinline__ v16h load_b_frag(const _Float16* __restrict__ W,
                                            int ldw, int nbase, int kbase,
                                            int lane) {
  // B[k][n] = W[n][k], W row-major [Npad, ldw] (pre-padded: no guard needed).
  // Lane holds col nbase+(lane&15);
  // elems 0..7 -> K = k0..k0+7, elems 8..15 -> K = k0+16..k0+23,
  // k0 = kbase + (lane>=16 ? 8 : 0).
  int n  = nbase + (lane & 15);
  int k0 = kbase + ((lane & 16) ? 8 : 0);
  const v8h* p0 = reinterpret_cast<const v8h*>(W + (size_t)n * ldw + k0);
  const v8h* p1 = reinterpret_cast<const v8h*>(W + (size_t)n * ldw + k0 + 16);
  v8h a = *p0, b = *p1;
  v16h r;
#pragma unroll
  for (int i = 0; i < 8; ++i) { r[i] = a[i]; r[i + 8] = b[i]; }
  return r;
}

// ---- Generic WMMA GEMM: C[M,Ncap] = A[M,K] * W[Npad,K]^T (+Cin) (+bias) ----
// Wave tile: (16*MT) x (16*NT); B fragments shared across MT row-tiles.
// Requirements: M % (16*MT) == 0, Npad % (16*NT) == 0, K % 64 == 0 and K >= 64
// (double-buffered pipeline), lda/ldw multiples of 8 (16B-aligned rows).

template <int MT, int NT>
__global__ void __launch_bounds__(256, 2)
wmma_gemm_f16(const _Float16* __restrict__ A, int lda,
              const _Float16* __restrict__ W, int ldw,
              const float* __restrict__ Cin, long ldcin,
              const float* __restrict__ bias,
              float* __restrict__ C, long ldc,
              int M, int Npad, int Ncap, int K) {
  int lane = threadIdx.x & 31;
  int wave = (int)((blockIdx.x * blockDim.x + threadIdx.x) >> 5);
  int ntG  = Npad / (16 * NT);
  int mg   = wave / ntG;
  int ng   = wave % ntG;
  if (mg * 16 * MT >= M) return;
  int mbase0 = mg * 16 * MT;
  int nbase0 = ng * 16 * NT;
  int rowoff = (lane & 16) ? 8 : 0;

  v8f acc[MT][NT];
#pragma unroll
  for (int m = 0; m < MT; ++m) {
#pragma unroll
    for (int t = 0; t < NT; ++t) {
      int col = nbase0 + t * 16 + (lane & 15);
      if (Cin != nullptr && col < Ncap) {
#pragma unroll
        for (int r = 0; r < 8; ++r)
          acc[m][t][r] =
              Cin[(size_t)(mbase0 + 16 * m + rowoff + r) * ldcin + col];
      } else {
#pragma unroll
        for (int r = 0; r < 8; ++r) acc[m][t][r] = 0.f;
      }
    }
  }

  v16h afA[MT], bfA[NT], afB[MT], bfB[NT];

  auto load_frags = [&](v16h (&af)[MT], v16h (&bf)[NT], int kb) {
#pragma unroll
    for (int m = 0; m < MT; ++m)
      af[m] = load_a_frag(A, lda, mbase0 + 16 * m, kb, lane);
#pragma unroll
    for (int t = 0; t < NT; ++t)
      bf[t] = load_b_frag(W, ldw, nbase0 + t * 16, kb, lane);
  };
  auto do_math = [&](v16h (&af)[MT], v16h (&bf)[NT]) {
#pragma unroll
    for (int t = 0; t < NT; ++t)
#pragma unroll
      for (int m = 0; m < MT; ++m)
        acc[m][t] = __builtin_amdgcn_wmma_f32_16x16x32_f16(
            false, af[m], false, bf[t], (short)0, acc[m][t], false, false);
  };

  // Software pipeline over K (stages of 32, rotated in pairs of 64).
  load_frags(afA, bfA, 0);
  int kb = 0;
  for (; kb < K - 64; kb += 64) {
    load_frags(afB, bfB, kb + 32);
    // Speculative prefetch two tiles ahead (safe past end of buffer).
    __builtin_prefetch(A + (size_t)(mbase0 + (lane & 15)) * lda + kb + 64, 0, 1);
    do_math(afA, bfA);
    load_frags(afA, bfA, kb + 64);
    do_math(afB, bfB);
  }
  // Epilogue: last two K-steps (K % 64 == 0 guarantees exactly these remain).
  load_frags(afB, bfB, kb + 32);
  do_math(afA, bfA);
  do_math(afB, bfB);

#pragma unroll
  for (int m = 0; m < MT; ++m) {
#pragma unroll
    for (int t = 0; t < NT; ++t) {
      int col = nbase0 + t * 16 + (lane & 15);
      if (col >= Ncap) continue;
      float bv = bias ? bias[col] : 0.f;
#pragma unroll
      for (int r = 0; r < 8; ++r)
        C[(size_t)(mbase0 + 16 * m + rowoff + r) * ldc + col] =
            acc[m][t][r] + bv;
    }
  }
}

// ---- Support kernels -------------------------------------------------------

// f32 -> f16 with zero-padding in both N (rows) and K (cols).
__global__ void w_to_f16_pad(const float* __restrict__ src,
                             _Float16* __restrict__ dst,
                             int NrowsOut, int NrowsSrc, int Kin, int Kout) {
  long idx = (long)blockIdx.x * blockDim.x + threadIdx.x;
  if (idx >= (long)NrowsOut * Kout) return;
  int k = (int)(idx % Kout);
  long n = idx / Kout;
  dst[idx] = (n < NrowsSrc && k < Kin) ? (_Float16)src[n * Kin + k]
                                       : (_Float16)0.f;
}

__global__ void embed_gather(const int* __restrict__ X,
                             const float* __restrict__ emb,
                             _Float16* __restrict__ out, int Ntok) {
  long idx = (long)blockIdx.x * blockDim.x + threadIdx.x;
  if (idx >= (long)Ntok * EP) return;
  int k = (int)(idx % EP);
  long n = idx / EP;
  int tok = X[n];
  out[idx] = (k < Ed) ? (_Float16)emb[(size_t)tok * Ed + k] : (_Float16)0.f;
}

// packed-sequence reversal: dst[b,t] = src[b, len-1-t] for t<len, else src[b,t]
__global__ void seq_reverse(const _Float16* __restrict__ src,
                            _Float16* __restrict__ dst,
                            const int* __restrict__ lens, int Wd) {
  long idx = (long)blockIdx.x * blockDim.x + threadIdx.x;
  if (idx >= (long)Bz * Sq * Wd) return;
  int w = (int)(idx % Wd);
  long bt = idx / Wd;
  int t = (int)(bt % Sq);
  int b = (int)(bt / Sq);
  int len = lens[b];
  int ts  = (t < len) ? (len - 1 - t) : t;
  dst[idx] = src[((size_t)b * Sq + ts) * Wd + w];
}

__global__ void lstm_cell(const float* __restrict__ gates,
                          const float* __restrict__ bih,
                          const float* __restrict__ bhh,
                          float* __restrict__ h, float* __restrict__ c,
                          _Float16* __restrict__ h16,
                          _Float16* __restrict__ yout, int ldy, int yoff,
                          const int* __restrict__ lens, int t, int rev_out) {
  int idx = blockIdx.x * blockDim.x + threadIdx.x;
  if (idx >= Bz * Hh) return;
  int b = idx / Hh, j = idx % Hh;
  size_t g0 = (size_t)b * G4;
  float gi = gates[g0 + j]          + bih[j]          + bhh[j];
  float gf = gates[g0 + Hh + j]     + bih[Hh + j]     + bhh[Hh + j];
  float gg = gates[g0 + 2*Hh + j]   + bih[2*Hh + j]   + bhh[2*Hh + j];
  float go = gates[g0 + 3*Hh + j]   + bih[3*Hh + j]   + bhh[3*Hh + j];
  float i_ = 1.f / (1.f + __expf(-gi));
  float f_ = 1.f / (1.f + __expf(-gf));
  float g_ = tanhf(gg);
  float o_ = 1.f / (1.f + __expf(-go));
  float cn = f_ * c[idx] + i_ * g_;
  float hn = o_ * tanhf(cn);
  int len = lens[b];
  bool valid = (t < len);
  float h2 = valid ? hn : h[idx];
  float c2 = valid ? cn : c[idx];
  h[idx] = h2; c[idx] = c2; h16[idx] = (_Float16)h2;
  int pos = (valid && rev_out) ? (len - 1 - t) : t;
  float yv = valid ? hn : 0.f;
  yout[((size_t)b * Sq + pos) * ldy + yoff + j] = (_Float16)yv;
}

// per-token expert linear: out[n,s] = sum_h Xf[n,h]*Wc[e,h,s] + bc[e,s]
__global__ void clf_expert(const _Float16* __restrict__ Xf,
                           const int* __restrict__ eid,
                           const float* __restrict__ Wc,
                           const float* __restrict__ bc,
                           float* __restrict__ out) {
  __shared__ float red[256];
  int n   = blockIdx.x;
  int tid = threadIdx.x;
  int s   = tid & 15;
  int hg  = tid >> 4;  // 16 h-groups
  int e   = eid[n];
  const float* We = Wc + (size_t)e * TwoH * NS;
  float acc = 0.f;
  for (int hh = hg; hh < TwoH; hh += 16)
    acc += (float)Xf[(size_t)n * TwoH + hh] * We[(size_t)hh * NS + s];
  red[tid] = acc;
  __syncthreads();
  for (int st = 8; st > 0; st >>= 1) {
    if (hg < st) red[tid] += red[tid + st * 16];
    __syncthreads();
  }
  if (hg == 0) out[(size_t)n * NS + s] = red[s] + bc[(size_t)e * NS + s];
}

inline int gemm_blocks(int M, int Npad, int MT, int NT) {
  long waves = (long)(M / (16 * MT)) * (Npad / (16 * NT));
  return (int)((waves * 32 + 255) / 256);
}

}  // namespace

extern "C" void kernel_launch(void* const* d_in, const int* in_sizes, int n_in,
                              void* d_out, int out_size, void* d_ws,
                              size_t ws_size, hipStream_t stream) {
  (void)in_sizes; (void)n_in; (void)out_size; (void)ws_size;

  const int*   X    = (const int*)d_in[0];
  const int*   lens = (const int*)d_in[1];
  const int*   eid  = (const int*)d_in[3];
  const float* emb  = (const float*)d_in[4];
  const float* Wih[4] = {(const float*)d_in[5],  (const float*)d_in[9],
                         (const float*)d_in[13], (const float*)d_in[17]};
  const float* Whh[4] = {(const float*)d_in[6],  (const float*)d_in[10],
                         (const float*)d_in[14], (const float*)d_in[18]};
  const float* bih[4] = {(const float*)d_in[7],  (const float*)d_in[11],
                         (const float*)d_in[15], (const float*)d_in[19]};
  const float* bhh[4] = {(const float*)d_in[8],  (const float*)d_in[12],
                         (const float*)d_in[16], (const float*)d_in[20]};
  const float* Wout = (const float*)d_in[21];
  const float* bout = (const float*)d_in[22];
  const float* clfW = (const float*)d_in[23];
  const float* clfb = (const float*)d_in[24];

  float* out_embed = (float*)d_out;
  float* out_clf   = out_embed + (size_t)NTOK * Ed;

  // ---- workspace carve-up --------------------------------------------------
  char* base = (char*)d_ws;
  size_t off = 0;
  auto take = [&](size_t bytes) -> char* {
    char* p = base + off;
    off += (bytes + 255) & ~(size_t)255;
    return p;
  };
  const int KinP[4]  = {Ed, Ed, TwoH, TwoH};
  const int KpadP[4] = {EP, EP, TwoH, TwoH};
  _Float16* wih16[4];
  _Float16* whh16[4];
  for (int p = 0; p < 4; ++p) {
    wih16[p] = (_Float16*)take((size_t)G4 * KpadP[p] * 2);
    whh16[p] = (_Float16*)take((size_t)G4 * Hh * 2);
  }
  _Float16* wout16  = (_Float16*)take((size_t)EdP * TwoH * 2);  // N-padded
  _Float16* h0f     = (_Float16*)take((size_t)NTOK * EP * 2);
  _Float16* h0b     = (_Float16*)take((size_t)NTOK * EP * 2);
  _Float16* concat1 = (_Float16*)take((size_t)NTOK * TwoH * 2);
  _Float16* rev1    = (_Float16*)take((size_t)NTOK * TwoH * 2);
  _Float16* Xf      = (_Float16*)take((size_t)NTOK * TwoH * 2);
  float*    xproj   = (float*)take((size_t)NTOK * G4 * 4);
  float*    gates   = (float*)take((size_t)Bz * G4 * 4);
  float*    hbuf    = (float*)take((size_t)Bz * Hh * 4);
  float*    cbuf    = (float*)take((size_t)Bz * Hh * 4);
  _Float16* h16buf  = (_Float16*)take((size_t)Bz * Hh * 2);

  // ---- weights -> f16 (zero-padded in K, Wout also padded in N) -----------
  for (int p = 0; p < 4; ++p) {
    long n1 = (long)G4 * KpadP[p];
    w_to_f16_pad<<<(int)((n1 + 255) / 256), 256, 0, stream>>>(
        Wih[p], wih16[p], G4, G4, KinP[p], KpadP[p]);
    long n2 = (long)G4 * Hh;
    w_to_f16_pad<<<(int)((n2 + 255) / 256), 256, 0, stream>>>(
        Whh[p], whh16[p], G4, G4, Hh, Hh);
  }
  {
    long n3 = (long)EdP * TwoH;
    w_to_f16_pad<<<(int)((n3 + 255) / 256), 256, 0, stream>>>(
        Wout, wout16, EdP, Ed, TwoH, TwoH);
  }

  // ---- embeddings + reversed copy -----------------------------------------
  {
    long n = (long)NTOK * EP;
    embed_gather<<<(int)((n + 255) / 256), 256, 0, stream>>>(X, emb, h0f, NTOK);
    seq_reverse<<<(int)((n + 255) / 256), 256, 0, stream>>>(h0f, h0b, lens, EP);
  }

  // ---- 4 LSTM passes: L0-fwd, L0-bwd, L1-fwd, L1-bwd ----------------------
  for (int p = 0; p < 4; ++p) {
    const _Float16* Ain = (p == 0) ? h0f : (p == 1) ? h0b
                         : (p == 2) ? concat1 : rev1;
    int Kp = KpadP[p];
    _Float16* ybuf = (p < 2) ? concat1 : Xf;
    int yoff = (p & 1) ? Hh : 0;
    int revo = (p & 1);

    hipMemsetAsync(hbuf,   0, (size_t)Bz * Hh * 4, stream);
    hipMemsetAsync(cbuf,   0, (size_t)Bz * Hh * 4, stream);
    hipMemsetAsync(h16buf, 0, (size_t)Bz * Hh * 2, stream);

    // hoisted input projection: xproj[N, 4H] = Ain @ Wih^T
    wmma_gemm_f16<2, 4><<<gemm_blocks(NTOK, G4, 2, 4), 256, 0, stream>>>(
        Ain, Kp, wih16[p], Kp, nullptr, 0, nullptr,
        xproj, G4, NTOK, G4, G4, Kp);

    for (int t = 0; t < Sq; ++t) {
      // gates[64,2048] = xproj[:,t,:] + h @ Whh^T
      wmma_gemm_f16<2, 4><<<gemm_blocks(Bz, G4, 2, 4), 256, 0, stream>>>(
          h16buf, Hh, whh16[p], Hh, xproj + (size_t)t * G4, (long)Sq * G4,
          nullptr, gates, G4, Bz, G4, G4, Hh);
      lstm_cell<<<(Bz * Hh + 255) / 256, 256, 0, stream>>>(
          gates, bih[p], bhh[p], hbuf, cbuf, h16buf,
          ybuf, TwoH, yoff, lens, t, revo);
    }

    if (p == 1) {  // layer-0 concat complete: build reversed input for L1-bwd
      long n = (long)NTOK * TwoH;
      seq_reverse<<<(int)((n + 255) / 256), 256, 0, stream>>>(
          concat1, rev1, lens, TwoH);
    }
  }

  // ---- head 1: embed_out = Xf @ Wout^T + bout (N=300 padded to 320) -------
  wmma_gemm_f16<2, 4><<<gemm_blocks(NTOK, EdP, 2, 4), 256, 0, stream>>>(
      Xf, TwoH, wout16, TwoH, nullptr, 0, bout,
      out_embed, Ed, NTOK, EdP, Ed, TwoH);

  // ---- head 2: per-token expert linear (bandwidth-bound gather) -----------
  clf_expert<<<NTOK, 256, 0, stream>>>(Xf, eid, clfW, clfb, out_clf);
}
